// SDFFunctionLayer_69423851372992
// MI455X (gfx1250) — matrix operand order repaired and checked by reading, not compiled, not run on hardware
//
#include <hip/hip_runtime.h>
#include <math.h>

// ---------------------------------------------------------------------------
// CDNA5 (gfx1250) fused SDF-MLP kernel.
//   h1 = tanh(rope @ W1 + b1)          GEMM1: M=2048 K=22(pad 32) N=256  (WMMA f16)
//   h2 = tanh(h1 @ W2 + b2)            GEMM2: M=2048 K=256       N=256  (WMMA f16)
//   point = h2 @ Wp + bp               folded into GEMM2 epilogue (shfl_xor reduce)
//   out = sigmoid(100*(-sdf[nn(point)] + bias_b))
//
// Roofline: ~0.3 GFLOP, live data (~700 KB) << 192 MB L2 -> latency-bound.
// 2048 rows / 16 rows-per-wave = 128 wave-tiles; 32 blocks x 4 waves saturate
// that exactly. All dataflow is wave-local (same-wave DS ops are in-order per
// the ISA), so the kernel needs NO barriers: every wave runs independently.
// ---------------------------------------------------------------------------

typedef _Float16 v16h __attribute__((ext_vector_type(16)));
typedef _Float16 v8h  __attribute__((ext_vector_type(8)));
typedef float    v8f  __attribute__((ext_vector_type(8)));

#define SDF_H      200
#define SDF_W      200
#define BATCH      2048
#define ROPE_DIM   22
#define K1PAD      32
#define FC         256
#define M_TILE     64          // batch rows per block (4 waves x 16 rows)
#define THREADS    128
#define LDS_STRIDE 264         // h1 row stride in f16 (256 + 8 pad)
#define SIG_SCALE  100.0f

// Branch-free tanh: CDNA5 hardware transcendental (v_tanh_f32) when available.
__device__ __forceinline__ float fast_tanh(float x) {
#if __has_builtin(__builtin_amdgcn_tanhf)
    return __builtin_amdgcn_tanhf(x);
#else
    // tanh(x) = 1 - 2/(exp2(2*log2e*x)+1); saturates correctly at +-inf.
    float e = __builtin_amdgcn_exp2f(x * 2.8853900817779268f);
    return 1.0f - 2.0f * __builtin_amdgcn_rcpf(e + 1.0f);
#endif
}

// ---------------------------------------------------------------------------
// Pre-pass: convert weights/activations to f16.
//   W1t[n][k]   K-major, k<32 zero-padded above 22        (256x32)
//   W2t[n][k]   K-major                                   (256x256)
//   ropeP[m][k] row-major f16, zero-padded 22->32         (2048x32)
// ---------------------------------------------------------------------------
__global__ void cvt_weights(const float* __restrict__ W1,
                            const float* __restrict__ W2,
                            const float* __restrict__ rope,
                            _Float16* __restrict__ W1t,
                            _Float16* __restrict__ W2t,
                            _Float16* __restrict__ ropeP) {
    int i = blockIdx.x * blockDim.x + threadIdx.x;
    if (i < FC * K1PAD) {
        int n = i / K1PAD, k = i % K1PAD;
        W1t[i] = (_Float16)((k < ROPE_DIM) ? W1[k * FC + n] : 0.0f);
    }
    if (i < FC * FC) {
        int n = i / FC, k = i % FC;
        W2t[i] = (_Float16)W2[k * FC + n];
    }
    if (i < BATCH * K1PAD) {
        int m = i / K1PAD, k = i % K1PAD;
        ropeP[i] = (_Float16)((k < ROPE_DIM) ? rope[m * ROPE_DIM + k] : 0.0f);
    }
}

// Build a 16-bit A fragment from two contiguous 16-byte runs (ISA layout:
// elements 0..7 -> K base..base+7, elements 8..15 -> K base+16..base+23).
__device__ __forceinline__ v16h load_frag_pair(const _Float16* p) {
    v8h lo = *(const v8h*)(p);
    v8h hi = *(const v8h*)(p + 16);
    v16h f;
#pragma unroll
    for (int e = 0; e < 8; ++e) { f[e] = lo[e]; f[e + 8] = hi[e]; }
    return f;
}

// ---------------------------------------------------------------------------
// Fused MLP + SDF lookup. One wave owns 16 batch rows end-to-end.
// ---------------------------------------------------------------------------
__global__ void sdf_mlp_kernel(const _Float16* __restrict__ ropeP,
                               const _Float16* __restrict__ W1t,
                               const float* __restrict__ b1,
                               const _Float16* __restrict__ W2t,
                               const float* __restrict__ b2,
                               const float* __restrict__ Wp,
                               const float* __restrict__ bp,
                               const float* __restrict__ bias_b,
                               const float* __restrict__ sdf_flat,
                               const float* __restrict__ sdf_res,
                               const float* __restrict__ sdf_org,
                               float* __restrict__ out) {
    extern __shared__ __align__(16) char smem_raw[];
    _Float16* h1     = (_Float16*)smem_raw;                                   // [M_TILE][LDS_STRIDE] f16
    float*    pointb = (float*)(smem_raw + (size_t)M_TILE * LDS_STRIDE * 2);  // [M_TILE][2]

    const int tid       = threadIdx.x;
    const int lane      = tid & 31;
    const int wv        = tid >> 5;          // wave id 0..3
    const int halfSel   = lane >> 4;         // 0: lanes 0-15, 1: lanes 16-31
    const int lm        = lane & 15;
    const int blockRow0 = blockIdx.x * M_TILE;
    const int waveRow0  = wv * 16;

    const int kBaseA = halfSel ? 8 : 0;      // A frag: K {0..7,16..23} / {8..15,24..31}
    const int kBaseB = halfSel ? 8 : 0;      // B frag: K {0..7,16..23} / {8..15,24..31}

    // ---------------- GEMM1: h1 = tanh(rope @ W1 + b1), K=32 (one WMMA/tile)
    const v16h a1 = load_frag_pair(ropeP + (size_t)(blockRow0 + waveRow0 + lm) * K1PAD + kBaseA);

#pragma unroll
    for (int t = 0; t < FC / 16; ++t) {
        const int n = t * 16 + lm;
        v16h bfrag = load_frag_pair(W1t + (size_t)n * K1PAD + kBaseB);
        v8f c = {};
        c = __builtin_amdgcn_wmma_f32_16x16x32_f16(false, a1, false, bfrag,
                                                   (short)0, c, false, false);
        const float bias = b1[n];
#pragma unroll
        for (int r = 0; r < 8; ++r) {
            // C/D layout: vgpr r -> row M = r + 8*halfSel, col N = lm
            int rowL = waveRow0 + halfSel * 8 + r;
            h1[rowL * LDS_STRIDE + t * 16 + lm] = (_Float16)fast_tanh(c[r] + bias);
        }
    }
    // No barrier: this wave only reads back its own 16 rows; same-wave DS ops
    // execute in order (ISA ch.11 / DScnt), so the store->load is ordered.

    // ---------------- GEMM2 (K=256) + fused point head
    // A fragments are N-invariant: load all 8 k-step fragments once (64 VGPRs).
    v16h aF[FC / 32];
    {
        const _Float16* ap = h1 + (waveRow0 + lm) * LDS_STRIDE + kBaseA;
#pragma unroll
        for (int ks = 0; ks < FC / 32; ++ks)
            aF[ks] = load_frag_pair(ap + ks * 32);
    }

    float px[8], py[8];
#pragma unroll
    for (int r = 0; r < 8; ++r) { px[r] = 0.0f; py[r] = 0.0f; }

#pragma unroll 4
    for (int t = 0; t < FC / 16; ++t) {
        const int n = t * 16 + lm;
        const _Float16* wp2 = W2t + (size_t)n * FC + kBaseB;
        v8f c = {};
#pragma unroll
        for (int ks = 0; ks < FC / 32; ++ks) {
            v16h bfrag = load_frag_pair(wp2 + ks * 32);
            c = __builtin_amdgcn_wmma_f32_16x16x32_f16(false, aF[ks], false, bfrag,
                                                       (short)0, c, false, false);
        }
        const float bias = b2[n];
        const float wx = Wp[n * 2 + 0];
        const float wy = Wp[n * 2 + 1];
#pragma unroll
        for (int r = 0; r < 8; ++r) {
            float hv = fast_tanh(c[r] + bias);  // h2(row waveRow0+8*halfSel+r, col n)
            px[r] += hv * wx;
            py[r] += hv * wy;
        }
    }
    // Reduce partial point sums across the 16 lanes of each half; every lane
    // of a half ends with the full sums for its 8 rows.
#pragma unroll
    for (int m = 1; m < 16; m <<= 1) {
#pragma unroll
        for (int r = 0; r < 8; ++r) {
            px[r] += __shfl_xor(px[r], m, 16);
            py[r] += __shfl_xor(py[r], m, 16);
        }
    }
    // Lane 0 stages rows 0..7, lane 16 rows 8..15 of this wave's tile (LDS,
    // wave-local, same-wave DS ordering again makes this safe without barrier).
    if (lm == 0) {
        const float bpx = bp[0], bpy = bp[1];
#pragma unroll
        for (int r = 0; r < 8; ++r) {
            int rowL = waveRow0 + halfSel * 8 + r;
            pointb[rowL * 2 + 0] = px[r] + bpx;
            pointb[rowL * 2 + 1] = py[r] + bpy;
        }
    }

    // ---------------- finalize: lanes 0..15 of each wave handle its 16 rows
    if (halfSel == 0) {
        const int rowL = waveRow0 + lm;
        const int g    = blockRow0 + rowL;
        float pxv = pointb[rowL * 2 + 0];
        float pyv = pointb[rowL * 2 + 1];
        float rx = sdf_res[g * 2 + 0], ry = sdf_res[g * 2 + 1];
        float ox = sdf_org[g * 2 + 0], oy = sdf_org[g * 2 + 1];
        int col = (int)rintf(pxv / rx + ox);   // rintf: round-half-even, matches jnp.round
        int row = (int)rintf(pyv / ry + oy);
        col = min(max(col, 0), SDF_W - 1);
        row = min(max(row, 0), SDF_H - 1);
        float sd = sdf_flat[(size_t)g * (SDF_H * SDF_W) + row * SDF_W + col];
        float logit = SIG_SCALE * (-sd + bias_b[0]);
        // branch-free native sigmoid: 1/(1+2^(-logit*log2e))
        float e = __builtin_amdgcn_exp2f(-1.4426950408889634f * logit);
        out[g] = __builtin_amdgcn_rcpf(1.0f + e);
    }
}

// ---------------------------------------------------------------------------
extern "C" void kernel_launch(void* const* d_in, const int* in_sizes, int n_in,
                              void* d_out, int out_size, void* d_ws, size_t ws_size,
                              hipStream_t stream) {
    const float* sdf_flat = (const float*)d_in[0];
    // d_in[1] = sdf_gradient_flat (655 MB): unused by the reference -> never touched
    const float* res    = (const float*)d_in[2];
    const float* org    = (const float*)d_in[3];
    const float* rope   = (const float*)d_in[4];
    const float* W1     = (const float*)d_in[5];
    const float* b1     = (const float*)d_in[6];
    const float* W2     = (const float*)d_in[7];
    const float* b2     = (const float*)d_in[8];
    const float* Wp     = (const float*)d_in[9];
    const float* bp     = (const float*)d_in[10];
    const float* bias_b = (const float*)d_in[11];
    float* out = (float*)d_out;

    _Float16* W1t   = (_Float16*)d_ws;                                  // 16 KB
    _Float16* W2t   = W1t + FC * K1PAD;                                 // 128 KB
    _Float16* ropeP = W2t + FC * FC;                                    // 128 KB

    cvt_weights<<<(BATCH * K1PAD + 255) / 256, 256, 0, stream>>>(W1, W2, rope,
                                                                 W1t, W2t, ropeP);

    size_t smem = (size_t)M_TILE * LDS_STRIDE * sizeof(_Float16)
                + (size_t)M_TILE * 2 * sizeof(float);
    sdf_mlp_kernel<<<BATCH / M_TILE, THREADS, smem, stream>>>(
        ropeP, W1t, b1, W2t, b2, Wp, bp, bias_b, sdf_flat, res, org, out);
}
